// MultiheadAttention_27676769255544
// MI455X (gfx1250) — matrix operand order repaired and checked by reading, not compiled, not run on hardware
//
#include <hip/hip_runtime.h>
#include <math.h>

// ---------------------------------------------------------------------------
// Problem constants (reference: B=2, L=2048, D=512, H=8, DK=DV=64)
// ---------------------------------------------------------------------------
#define BB   2
#define LL   2048
#define DD   512
#define HH   8
#define DKV  64
#define MROWS (BB * LL)        // 4096
#define EPS_LN 1e-6f

typedef __attribute__((ext_vector_type(16))) _Float16 v16h;
typedef __attribute__((ext_vector_type(8)))  float    v8f;
typedef __attribute__((ext_vector_type(2)))  _Float16 h2;
typedef __attribute__((ext_vector_type(8)))  _Float16 h8;

// ---------------------------------------------------------------------------
// Helpers
// ---------------------------------------------------------------------------
__device__ __forceinline__ v8f wmma_f16(v16h a, v16h b, v8f c) {
  return __builtin_amdgcn_wmma_f32_16x16x32_f16(
      false, a, false, b, (short)0, c, false, false);
}

__device__ __forceinline__ v16h vneg16(v16h a) {
  v16h r;
#pragma unroll
  for (int e = 0; e < 16; ++e) r[e] = -a[e];
  return r;
}

__device__ __forceinline__ h2 pk16(float a, float b) {
  // v_cvt_pk_rtz_f16_f32 ; builtin returns __fp16x2 -> bit-cast to _Float16x2
  return __builtin_bit_cast(h2, __builtin_amdgcn_cvt_pkrtz(a, b));
}

// CDNA5 async copy: global -> LDS, 16B per lane, tracked with ASYNCcnt.
__device__ __forceinline__ void async_copy_b128(uint32_t lds_off,
                                                const _Float16* gsrc) {
  asm volatile("global_load_async_to_lds_b128 %0, %1, off"
               :: "v"(lds_off), "v"(gsrc) : "memory");
}
__device__ __forceinline__ void wait_async0() {
  asm volatile("s_wait_asynccnt 0" ::: "memory");
}

// A-fragment (16x32 f16, MxK) from LDS tile [m][k], stride ldk halves.
// lane L -> M = L%16; element e -> K = (e/8)*16 + (L/16)*8 + (e%8).
__device__ __forceinline__ v16h load_a_frag(const _Float16* lds, int m_base,
                                            int ldk, int lane) {
  int m = lane & 15, half = lane >> 4;
  const _Float16* rp = lds + (m_base + m) * ldk + half * 8;
  v16h a;
#pragma unroll
  for (int e = 0; e < 16; ++e) a[e] = rp[(e >> 3) * 16 + (e & 7)];
  return a;
}

// B-fragment (32x16 f16, KxN) from LDS holding the transposed operand [n][k].
// lane L -> N = L%16; element e -> K = (e/8)*16 + (L/16)*8 + (e%8).
__device__ __forceinline__ v16h load_b_frag(const _Float16* lds, int n_base,
                                            int ldk, int k_base, int lane) {
  int n = lane & 15, kh = lane >> 4;
  const _Float16* rp = lds + (n_base + n) * ldk + k_base + kh * 8;
  v16h b;
#pragma unroll
  for (int e = 0; e < 16; ++e) b[e] = rp[(e >> 3) * 16 + (e & 7)];
  return b;
}

// ---------------------------------------------------------------------------
// Kernel 1: complex GEMM  Y = scale * (X @ W^T), X:[M,K] (TX), W:[N,K] fp32,
// Y:[M,N] (TY).  Block 256 = 8 waves -> 32x64 tile (2x4 waves of 16x16).
// ---------------------------------------------------------------------------
template <typename TX, typename TY>
__global__ __launch_bounds__(256) void cgemm_kernel(
    const TX* __restrict__ Xr, const TX* __restrict__ Xi,
    const float* __restrict__ Wr, const float* __restrict__ Wi,
    TY* __restrict__ Yr, TY* __restrict__ Yi,
    int M, int N, int K, float scale) {
  __shared__ _Float16 sXr[32][40], sXi[32][40];
  __shared__ _Float16 sWr[64][40], sWi[64][40];

  const int tid  = threadIdx.x;
  const int wave = tid >> 5, lane = tid & 31;
  const int mblk = blockIdx.y * 32, nblk = blockIdx.x * 64;
  const int msub = (wave >> 2) * 16, nsub = (wave & 3) * 16;

  v8f accr = {}, acci = {};

  for (int k0 = 0; k0 < K; k0 += 32) {
    __syncthreads();
    if constexpr (sizeof(TX) == 4) {
      // fp32 X: packed convert, one b32 LDS store per float pair
#pragma unroll
      for (int it = 0; it < 2; ++it) {
        int idx = tid + it * 256;              // 512 pairs per matrix
        int r = idx >> 4, c2 = (idx & 15) * 2;
        size_t g = (size_t)(mblk + r) * K + k0 + c2;
        *(h2*)&sXr[r][c2] = pk16(Xr[g], Xr[g + 1]);
        *(h2*)&sXi[r][c2] = pk16(Xi[g], Xi[g + 1]);
      }
    } else {
      // f16 X: straight 16B copies (128 chunks/matrix, 256 threads)
      int mat = tid >> 7, c = tid & 127;
      int r = c >> 2, c8 = (c & 3) * 8;
      const _Float16* src = mat ? (const _Float16*)Xi : (const _Float16*)Xr;
      _Float16* dst = mat ? &sXi[0][0] : &sXr[0][0];
      size_t g = (size_t)(mblk + r) * K + k0 + c8;
      *(h8*)&dst[r * 40 + c8] = *(const h8*)&src[g];
    }
#pragma unroll
    for (int it = 0; it < 4; ++it) {           // W tile 64x32, 1024 pairs
      int idx = tid + it * 256;
      int r = idx >> 4, c2 = (idx & 15) * 2;
      size_t g = (size_t)(nblk + r) * K + k0 + c2;
      *(h2*)&sWr[r][c2] = pk16(Wr[g], Wr[g + 1]);
      *(h2*)&sWi[r][c2] = pk16(Wi[g], Wi[g + 1]);
    }
    __syncthreads();

    v16h ar = load_a_frag(&sXr[0][0], msub, 40, lane);
    v16h ai = load_a_frag(&sXi[0][0], msub, 40, lane);
    v16h br = load_b_frag(&sWr[0][0], nsub, 40, 0, lane);
    v16h bi = load_b_frag(&sWi[0][0], nsub, 40, 0, lane);
    v16h ain = vneg16(ai);
    accr = wmma_f16(ar,  br, accr);
    accr = wmma_f16(ain, bi, accr);
    acci = wmma_f16(ar,  bi, acci);
    acci = wmma_f16(ai,  br, acci);
  }

  const int half = lane >> 4, n = lane & 15;
#pragma unroll
  for (int j = 0; j < 8; ++j) {                // C layout: M=j+8*half, N=lane%16
    size_t o = (size_t)(mblk + msub + j + 8 * half) * N + (nblk + nsub + n);
    if constexpr (sizeof(TY) == 4) {
      Yr[o] = accr[j] * scale;
      Yi[o] = acci[j] * scale;
    } else {
      Yr[o] = (TY)(accr[j] * scale);
      Yi[o] = (TY)(acci[j] * scale);
    }
  }
}

// ---------------------------------------------------------------------------
// Kernel 2: complex flash attention with CVSoftMax (all f16 operands).
// Grid: (L/128, B*H). Block 256 = 8 waves; wave owns 16 query rows.
// 32 keys/iteration; K tiles staged via global_load_async_to_lds_b128.
// ---------------------------------------------------------------------------
__global__ __launch_bounds__(256) void cattn_kernel(
    const _Float16* __restrict__ qhr, const _Float16* __restrict__ qhi,
    const _Float16* __restrict__ khr, const _Float16* __restrict__ khi,
    const _Float16* __restrict__ vhr, const _Float16* __restrict__ vhi,
    _Float16* __restrict__ outr, _Float16* __restrict__ outi) {
  __shared__ _Float16 sKr[32][72],  sKi[32][72];      // [key][d]
  __shared__ _Float16 sVtr[64][40], sVti[64][40];     // [dv][key]
  __shared__ _Float16 sPr[8][16][40], sPi[8][16][40]; // per-wave P-hat [m][key]

  const int tid  = threadIdx.x;
  const int wave = tid >> 5, lane = tid & 31;
  const int b = blockIdx.y / HH, h = blockIdx.y % HH;
  const int qbase = blockIdx.x * 128 + wave * 16;
  const int m = lane & 15, halfw = lane >> 4;

  // --- preload Q fragments directly from global f16 (vector 16B chunks) ---
  const _Float16* qpr = qhr + (size_t)(b * LL + qbase + m) * DD + h * DKV;
  const _Float16* qpi = qhi + (size_t)(b * LL + qbase + m) * DD + h * DKV;
  v16h qar[2], qai[2], qain[2];
#pragma unroll
  for (int kk = 0; kk < 2; ++kk) {
#pragma unroll
    for (int grp = 0; grp < 2; ++grp) {
      h8 cr = *(const h8*)&qpr[kk * 32 + grp * 16 + halfw * 8];
      h8 ci = *(const h8*)&qpi[kk * 32 + grp * 16 + halfw * 8];
#pragma unroll
      for (int e = 0; e < 8; ++e) { qar[kk][grp * 8 + e] = cr[e];
                                    qai[kk][grp * 8 + e] = ci[e]; }
    }
    qain[kk] = vneg16(qai[kk]);
  }

  v8f oaccr[4], oacci[4];
  v8f vz = {};
#pragma unroll
  for (int t = 0; t < 4; ++t) { oaccr[t] = vz; oacci[t] = vz; }
  float mrun[8], lrun[8];
#pragma unroll
  for (int j = 0; j < 8; ++j) { mrun[j] = -1e30f; lrun[j] = 0.0f; }

  for (int kb = 0; kb < LL / 32; ++kb) {
    __syncthreads();
    // ---- K tile: async global->LDS copy (one 16B chunk per thread/matrix)
    {
      int r = tid >> 3, c8 = (tid & 7) * 8;    // 32 rows x 8 chunks
      size_t g = (size_t)(b * LL + kb * 32 + r) * DD + h * DKV + c8;
      async_copy_b128((uint32_t)(uintptr_t)&sKr[r][c8], &khr[g]);
      async_copy_b128((uint32_t)(uintptr_t)&sKi[r][c8], &khi[g]);
    }
    // ---- V tile: transposed store [dv][key]
    {
      int r = tid >> 3, c8 = (tid & 7) * 8;
      size_t g = (size_t)(b * LL + kb * 32 + r) * DD + h * DKV + c8;
      h8 vr = *(const h8*)&vhr[g];
      h8 vi = *(const h8*)&vhi[g];
#pragma unroll
      for (int e = 0; e < 8; ++e) { sVtr[c8 + e][r] = vr[e];
                                    sVti[c8 + e][r] = vi[e]; }
    }
    wait_async0();
    __syncthreads();

    // ---- S = Q @ K^T (complex), two 16x16 n-subtiles ----
    v8f sr[2], si[2];
#pragma unroll
    for (int ns = 0; ns < 2; ++ns) {
      v8f cr = {}, ci = {};
#pragma unroll
      for (int kk = 0; kk < 2; ++kk) {
        v16h br = load_b_frag(&sKr[0][0], ns * 16, 72, kk * 32, lane);
        v16h bi = load_b_frag(&sKi[0][0], ns * 16, 72, kk * 32, lane);
        cr = wmma_f16(qar[kk],  br, cr);
        cr = wmma_f16(qain[kk], bi, cr);
        ci = wmma_f16(qar[kk],  bi, ci);
        ci = wmma_f16(qai[kk],  br, ci);
      }
      sr[ns] = cr; si[ns] = ci;
    }

    // ---- online CVSoftMax: softmax over |S|, phase preserved ----
    // |s| = m2 * rsq(m2);  p/|s| = p * rsq(m2)  (one v_rsq_f32 per element)
    float alpha[8];
#pragma unroll
    for (int j = 0; j < 8; ++j) {              // row M = j + 8*halfw
      float m2_0 = sr[0][j] * sr[0][j] + si[0][j] * si[0][j] + 1e-30f;
      float m2_1 = sr[1][j] * sr[1][j] + si[1][j] * si[1][j] + 1e-30f;
      float rq0 = __builtin_amdgcn_rsqf(m2_0);
      float rq1 = __builtin_amdgcn_rsqf(m2_1);
      float mag0 = m2_0 * rq0, mag1 = m2_1 * rq1;
      float lm = fmaxf(mag0, mag1);
#pragma unroll
      for (int sft = 8; sft >= 1; sft >>= 1) lm = fmaxf(lm, __shfl_xor(lm, sft, 32));
      float nm = fmaxf(mrun[j], lm);
      float a  = __expf(mrun[j] - nm);
      float p0 = __expf(mag0 - nm), p1 = __expf(mag1 - nm);
      float rs = p0 + p1;
#pragma unroll
      for (int sft = 8; sft >= 1; sft >>= 1) rs += __shfl_xor(rs, sft, 32);
      lrun[j] = a * lrun[j] + rs;
      mrun[j] = nm;
      alpha[j] = a;
      float i0 = p0 * rq0;
      float i1 = p1 * rq1;
      int row = j + 8 * halfw;
      sPr[wave][row][m]      = (_Float16)(sr[0][j] * i0);
      sPi[wave][row][m]      = (_Float16)(si[0][j] * i0);
      sPr[wave][row][16 + m] = (_Float16)(sr[1][j] * i1);
      sPi[wave][row][16 + m] = (_Float16)(si[1][j] * i1);
    }
#pragma unroll
    for (int t = 0; t < 4; ++t)
#pragma unroll
      for (int j = 0; j < 8; ++j) { oaccr[t][j] *= alpha[j]; oacci[t][j] *= alpha[j]; }

    // ---- O += P-hat @ V (complex); same-wave LDS, DS ops in order ----
    v16h par  = load_a_frag(&sPr[wave][0][0], 0, 40, lane);
    v16h pai  = load_a_frag(&sPi[wave][0][0], 0, 40, lane);
    v16h pain = vneg16(pai);
#pragma unroll
    for (int t = 0; t < 4; ++t) {
      v16h br = load_b_frag(&sVtr[0][0], t * 16, 40, 0, lane);
      v16h bi = load_b_frag(&sVti[0][0], t * 16, 40, 0, lane);
      oaccr[t] = wmma_f16(par,  br, oaccr[t]);
      oaccr[t] = wmma_f16(pain, bi, oaccr[t]);
      oacci[t] = wmma_f16(par,  bi, oacci[t]);
      oacci[t] = wmma_f16(pai,  br, oacci[t]);
    }
  }

  // ---- normalize by row sums; store f16 ----
#pragma unroll
  for (int t = 0; t < 4; ++t)
#pragma unroll
    for (int j = 0; j < 8; ++j) {
      size_t row = (size_t)(b * LL + qbase + j + 8 * halfw);
      size_t o = row * DD + h * DKV + t * 16 + m;
      float inv = __builtin_amdgcn_rcpf(fmaxf(lrun[j], 1e-30f));
      outr[o] = (_Float16)(oaccr[t][j] * inv);
      outi[o] = (_Float16)(oacci[t][j] * inv);
    }
}

// ---------------------------------------------------------------------------
// Kernel 3: residual + covariance-whitening complex LayerNorm + affine.
// ---------------------------------------------------------------------------
__device__ __forceinline__ float block_sum(float v, float* red, int tid) {
#pragma unroll
  for (int s = 16; s >= 1; s >>= 1) v += __shfl_xor(v, s, 32);
  __syncthreads();
  if ((tid & 31) == 0) red[tid >> 5] = v;
  __syncthreads();
  float t = red[0];
#pragma unroll
  for (int w = 1; w < 8; ++w) t += red[w];
  return t;
}

__global__ __launch_bounds__(256) void resln_kernel(
    const float* __restrict__ fr, const float* __restrict__ fi,
    const float* __restrict__ qr, const float* __restrict__ qi,
    const float* __restrict__ lnw, const float* __restrict__ lnbr,
    const float* __restrict__ lnbi, float* __restrict__ out) {
  __shared__ float red[8];
  const int tid = threadIdx.x;
  const size_t base = (size_t)blockIdx.x * DD;
  const int d0 = tid, d1 = tid + 256;

  float xr0 = fr[base + d0] + qr[base + d0];
  float xi0 = fi[base + d0] + qi[base + d0];
  float xr1 = fr[base + d1] + qr[base + d1];
  float xi1 = fi[base + d1] + qi[base + d1];

  const float rD = 1.0f / (float)DD;
  float mr = block_sum(xr0 + xr1, red, tid) * rD;
  float mi = block_sum(xi0 + xi1, red, tid) * rD;
  xr0 -= mr; xr1 -= mr; xi0 -= mi; xi1 -= mi;

  float vrr = block_sum(xr0 * xr0 + xr1 * xr1, red, tid) * rD + EPS_LN;
  float vii = block_sum(xi0 * xi0 + xi1 * xi1, red, tid) * rD + EPS_LN;
  float vri = block_sum(xr0 * xi0 + xr1 * xi1, red, tid) * rD;

  float s   = __builtin_amdgcn_sqrtf(vrr * vii - vri * vri);
  float t   = __builtin_amdgcn_sqrtf(vrr + vii + 2.0f * s);
  float inv = __builtin_amdgcn_rcpf(s * t);
  float rrr = (vii + s) * inv, rii = (vrr + s) * inv, rri = -vri * inv;

  float yr0 = rrr * xr0 + rri * xi0, yi0 = rri * xr0 + rii * xi0;
  float yr1 = rrr * xr1 + rri * xi1, yi1 = rri * xr1 + rii * xi1;

  out[(base + d0) * 2 + 0] = lnw[d0] * yr0 + lnw[512 + d0] * yi0 + lnbr[d0];
  out[(base + d0) * 2 + 1] = lnw[1024 + d0] * yr0 + lnw[1536 + d0] * yi0 + lnbi[d0];
  out[(base + d1) * 2 + 0] = lnw[d1] * yr1 + lnw[512 + d1] * yi1 + lnbr[d1];
  out[(base + d1) * 2 + 1] = lnw[1024 + d1] * yr1 + lnw[1536 + d1] * yi1 + lnbi[d1];
}

// ---------------------------------------------------------------------------
// Host launcher
// ---------------------------------------------------------------------------
extern "C" void kernel_launch(void* const* d_in, const int* in_sizes, int n_in,
                              void* d_out, int out_size, void* d_ws, size_t ws_size,
                              hipStream_t stream) {
  (void)in_sizes; (void)n_in; (void)out_size; (void)ws_size;
  const float* q_r  = (const float*)d_in[0];
  const float* q_i  = (const float*)d_in[1];
  const float* k_r  = (const float*)d_in[2];
  const float* k_i  = (const float*)d_in[3];
  const float* v_r  = (const float*)d_in[4];
  const float* v_i  = (const float*)d_in[5];
  const float* wq_r = (const float*)d_in[6];
  const float* wq_i = (const float*)d_in[7];
  const float* wk_r = (const float*)d_in[8];
  const float* wk_i = (const float*)d_in[9];
  const float* wv_r = (const float*)d_in[10];
  const float* wv_i = (const float*)d_in[11];
  const float* fc_r = (const float*)d_in[12];
  const float* fc_i = (const float*)d_in[13];
  const float* ln_w  = (const float*)d_in[14];
  const float* ln_br = (const float*)d_in[15];
  const float* ln_bi = (const float*)d_in[16];

  // Workspace: 8 f16 buffers [4096x512] (4 MB each) + 2 fp32 buffers (8 MB).
  const size_t NEL   = (size_t)MROWS * DD;
  const size_t F16SZ = NEL * sizeof(_Float16);
  char* wsb = (char*)d_ws;
  _Float16* qhr = (_Float16*)(wsb + 0 * F16SZ);
  _Float16* qhi = (_Float16*)(wsb + 1 * F16SZ);
  _Float16* khr = (_Float16*)(wsb + 2 * F16SZ);
  _Float16* khi = (_Float16*)(wsb + 3 * F16SZ);
  _Float16* vhr = (_Float16*)(wsb + 4 * F16SZ);
  _Float16* vhi = (_Float16*)(wsb + 5 * F16SZ);
  _Float16* aor = (_Float16*)(wsb + 6 * F16SZ);
  _Float16* aoi = (_Float16*)(wsb + 7 * F16SZ);
  float*    fcr = (float*)(wsb + 8 * F16SZ);
  float*    fci = fcr + NEL;

  dim3 gG(DD / 64, MROWS / 32);    // (8, 128)
  const float qscale = 0.125f;     // 1/sqrt(DK)
  hipLaunchKernelGGL((cgemm_kernel<float, _Float16>), gG, dim3(256), 0, stream,
                     q_r, q_i, wq_r, wq_i, qhr, qhi, MROWS, DD, DD, qscale);
  hipLaunchKernelGGL((cgemm_kernel<float, _Float16>), gG, dim3(256), 0, stream,
                     k_r, k_i, wk_r, wk_i, khr, khi, MROWS, DD, DD, 1.0f);
  hipLaunchKernelGGL((cgemm_kernel<float, _Float16>), gG, dim3(256), 0, stream,
                     v_r, v_i, wv_r, wv_i, vhr, vhi, MROWS, DD, DD, 1.0f);

  dim3 gA(LL / 128, BB * HH);      // (16, 16)
  hipLaunchKernelGGL(cattn_kernel, gA, dim3(256), 0, stream,
                     qhr, qhi, khr, khi, vhr, vhi, aor, aoi);

  hipLaunchKernelGGL((cgemm_kernel<_Float16, float>), gG, dim3(256), 0, stream,
                     aor, aoi, fc_r, fc_i, fcr, fci, MROWS, DD, DD, 1.0f);

  hipLaunchKernelGGL(resln_kernel, dim3(MROWS), dim3(256), 0, stream,
                     fcr, fci, q_r, q_i, ln_w, ln_br, ln_bi, (float*)d_out);
}
